// LatentSDE_28114855920151
// MI455X (gfx1250) — matrix operand order repaired and checked by reading, not compiled, not run on hardware
//
#include <hip/hip_runtime.h>
#include <hip/hip_bf16.h>

// ---------------------------------------------------------------------------
// Latent SDE for MI455X (gfx1250, wave32, WMMA).
// Phase 0: parallel weight repack f32 -> f16, B matrices pre-swizzled into
//          WMMA fragment order (each lane's 16 halves contiguous -> ds_load_b128).
// Phase 1: persistent single-workgroup backward GRU + ctx projection (WMMA).
// Phase 2: persistent single-workgroup SDE scan: f/h MLPs via WMMA, g elementwise.
// Phase 3: wide parallel Gaussian log-prob reduction.
// ---------------------------------------------------------------------------

typedef __attribute__((ext_vector_type(16))) _Float16 v16h;
typedef __attribute__((ext_vector_type(8)))  _Float16 v8h;
typedef __attribute__((ext_vector_type(8)))  float    v8f;

#define T_STEPS 1024
#define BATCH   256
#define DIN     16
#define CDIM    64
#define HDIM    128
#define OUTD    8
#define G3H     384   // 3*HDIM

__device__ __forceinline__ float softplus_f(float x) {
    float ax = fabsf(x);
    return fmaxf(x, 0.f) + __logf(1.f + __expf(-ax));
}
__device__ __forceinline__ float sigmoid_f(float x) {
    return 1.f / (1.f + __expf(-x));
}
__device__ __forceinline__ float tanh_f(float x) {
    float e = __expf(2.f * x);
    return (e - 1.f) / (e + 1.f);
}

// A fragment: 16x32 f16 tile of row-major A (lda elems/row) at (row0,k0).
// Two 16-byte LDS loads per lane -> ds_load_b128 x2.
__device__ __forceinline__ v16h load_a_frag(const _Float16* __restrict__ A, int lda,
                                            int row0, int k0, int lane) {
    int m  = lane & 15;
    int kh = (lane >> 4) << 3;
    const _Float16* p = A + (row0 + m) * lda + k0 + kh;
    v8h lo = *(const v8h*)p;
    v8h hi = *(const v8h*)(p + 16);
    return __builtin_shufflevector(lo, hi, 0, 1, 2, 3, 4, 5, 6, 7,
                                   8, 9, 10, 11, 12, 13, 14, 15);
}

// B fragment from pre-swizzled storage: tile-major, lane-major, 16 contiguous halves.
__device__ __forceinline__ v16h load_b_sw(const _Float16* __restrict__ Bf, int tile, int lane) {
    return *(const v16h*)(Bf + (((tile << 5) + lane) << 4));
}

// Generic 256xN = 256xK @ KxN layer, bias + optional softplus, f16 output to LDS.
// Bf is the fragment-swizzled weight; tiles indexed (kb * (N/16) + nt).
template <int KBLOCKS, int ACT>
__device__ __forceinline__ void gemm_layer(const _Float16* __restrict__ A, int lda, int k0A,
                                           const _Float16* __restrict__ Bf, int N,
                                           const float* __restrict__ bias,
                                           _Float16* __restrict__ outH, int ldo,
                                           int lane, int wave, int nwaves) {
    const int ntt    = N >> 4;
    const int ntasks = (BATCH >> 4) * ntt;
    for (int task = wave; task < ntasks; task += nwaves) {
        int mt = task / ntt, nt = task - mt * ntt;
        int row0 = mt << 4, col0 = nt << 4;
        int col = col0 + (lane & 15);
        float bv = bias[col];
        v8f acc;
#pragma unroll
        for (int r = 0; r < 8; ++r) acc[r] = bv;
#pragma unroll
        for (int kb = 0; kb < KBLOCKS; ++kb) {
            v16h a = load_a_frag(A, lda, row0, k0A + kb * 32, lane);
            v16h b = load_b_sw(Bf, kb * ntt + nt, lane);
            acc = __builtin_amdgcn_wmma_f32_16x16x32_f16(false, a, false, b, (short)0, acc,
                                                         false, false);
        }
        int rb = row0 + ((lane >> 4) << 3);
#pragma unroll
        for (int r = 0; r < 8; ++r) {
            float v = acc[r];
            if (ACT == 1) v = softplus_f(v);
            outH[(rb + r) * ldo + col] = (_Float16)v;
        }
    }
}

// vectorized global->LDS f16 copy (8 halves per move)
__device__ __forceinline__ void copy_h8(const _Float16* __restrict__ src,
                                        _Float16* __restrict__ dst,
                                        int n_halves, int tid, int nthr) {
    const uint4* s = (const uint4*)src;
    uint4* d = (uint4*)dst;
    for (int i = tid; i < (n_halves >> 3); i += nthr) d[i] = s[i];
}

// ------------------------------ weight prep --------------------------------
// Pack B = W^T (W is (N,K) row-major), zero-padded to Kp rows, into WMMA
// fragment order: out[((kb*(N/16)+nt)*32 + lane)*16 + i] =
//   Wt[kb*32 + (lane>>4)*8 + (i&7) + (i>>3)*16][nt*16 + (lane&15)].
__global__ void pack_b_frag(const float* __restrict__ W, _Float16* __restrict__ out,
                            int N, int K, int Kp) {
    int idx = blockIdx.x * 256 + threadIdx.x;
    int total = Kp * N;
    if (idx >= total) return;
    int i    = idx & 15;
    int lane = (idx >> 4) & 31;
    int tile = idx >> 9;
    int ntt  = N >> 4;
    int kb   = tile / ntt, nt = tile - kb * ntt;
    int k = kb * 32 + ((lane >> 4) << 3) + (i & 7) + ((i >> 3) << 4);
    int n = nt * 16 + (lane & 15);
    out[idx] = (_Float16)((k < K) ? W[n * K + k] : 0.0f);
}

// f_W1 is (128, 80) over concat([y(16), c(64)]). Staged A is [c | y | pad] (K=96):
// k<64 -> j=16+k (c), 64<=k<80 -> j=k-64 (y), else 0. Fragment-swizzled output.
__global__ void pack_fW1_frag(const float* __restrict__ W, _Float16* __restrict__ out) {
    int idx = blockIdx.x * 256 + threadIdx.x;
    if (idx >= 96 * HDIM) return;
    int i    = idx & 15;
    int lane = (idx >> 4) & 31;
    int tile = idx >> 9;
    int kb = tile >> 3, nt = tile & 7;     // ntt = 128/16 = 8
    int k = kb * 32 + ((lane >> 4) << 3) + (i & 7) + ((i >> 3) << 4);
    int n = nt * 16 + (lane & 15);
    float v = 0.f;
    if (k < 64)      v = W[n * 80 + 16 + k];
    else if (k < 80) v = W[n * 80 + (k - 64)];
    out[idx] = (_Float16)v;
}

__global__ void conv_f16_kernel(const float* __restrict__ in, _Float16* __restrict__ out, int n) {
    int idx = blockIdx.x * 256 + threadIdx.x;
    if (idx < n) out[idx] = (_Float16)in[idx];
}

// --------------------------- Phase 1: GRU + ctx ----------------------------
extern "C" __global__ __launch_bounds__(512)
void gru_ctx_kernel(const float* __restrict__ xs,
                    const float* __restrict__ bih, const float* __restrict__ bhh,
                    const float* __restrict__ encb,
                    const _Float16* __restrict__ wWiT, const _Float16* __restrict__ wWhT,
                    const _Float16* __restrict__ weT,
                    _Float16* __restrict__ ctx) {
    extern __shared__ __attribute__((aligned(32))) char smem[];
    _Float16* sWi = (_Float16*)smem;         // 32 x 384  (swizzled, ntt=24)
    _Float16* sWh = sWi + 32 * G3H;          // 128 x 384 (swizzled, ntt=24)
    _Float16* seT = sWh + HDIM * G3H;        // 128 x 64  (swizzled, ntt=4)
    _Float16* sx  = seT + HDIM * CDIM;       // 256 x 32  (x padded, row-major)
    _Float16* sh0 = sx + BATCH * 32;         // 256 x 128
    _Float16* sh1 = sh0 + BATCH * HDIM;      // 256 x 128
    float* sbih  = (float*)(sh1 + BATCH * HDIM); // 384
    float* sbhh  = sbih + G3H;                   // 384
    float* sencb = sbhh + G3H;                   // 64

    const int tid = threadIdx.x, lane = tid & 31, wave = tid >> 5;
    const int NW = 16;

    copy_h8(wWiT, sWi, 32 * G3H, tid, 512);
    copy_h8(wWhT, sWh, HDIM * G3H, tid, 512);
    copy_h8(weT, seT, HDIM * CDIM, tid, 512);
    for (int i = tid; i < G3H; i += 512) { sbih[i] = bih[i]; sbhh[i] = bhh[i]; }
    for (int i = tid; i < CDIM; i += 512) sencb[i] = encb[i];
    for (int i = tid; i < BATCH * HDIM; i += 512) sh0[i] = (_Float16)0.f;  // h0 = 0
    for (int i = tid; i < BATCH * 32; i += 512) sx[i] = (_Float16)0.f;     // pad cols
    __syncthreads();

    for (int s = 0; s < T_STEPS; ++s) {
        const int t = T_STEPS - 1 - s;              // backward scan
        const _Float16* hcur = (s & 1) ? sh1 : sh0;
        _Float16*       hnxt = (s & 1) ? sh0 : sh1;

        const float* xp = xs + (size_t)t * BATCH * DIN;
        for (int i = tid; i < BATCH * DIN; i += 512) {
            int b = i >> 4, d = i & 15;
            sx[b * 32 + d] = (_Float16)xp[i];
        }
        if (s + 1 < T_STEPS && tid < 256) {         // prefetch next x row
            const char* p = (const char*)(xs + (size_t)(t - 1) * BATCH * DIN);
            __builtin_prefetch(p + tid * 64, 0, 1);
        }
        __syncthreads();

        // fused GRU gate GEMMs: tasks over (m-tile 0..15, h-tile 0..7)
        for (int task = wave; task < 128; task += NW) {
            int mt = task >> 3, ht = task & 7;
            int row0 = mt << 4, col0 = ht << 4;
            int col = col0 + (lane & 15);
            v16h ax = load_a_frag(sx, 32, row0, 0, lane);
            v16h ah[4];
#pragma unroll
            for (int kb = 0; kb < 4; ++kb) ah[kb] = load_a_frag(hcur, HDIM, row0, kb * 32, lane);
            v8f gi[3], gh[3];
#pragma unroll
            for (int g = 0; g < 3; ++g) {
                int ntg = g * 8 + ht;               // column tile in 384-wide output
                v8f ai, ahh;
#pragma unroll
                for (int r = 0; r < 8; ++r) { ai[r] = 0.f; ahh[r] = 0.f; }
                v16h bw = load_b_sw(sWi, ntg, lane);                  // ntt=24, kb=0
                ai = __builtin_amdgcn_wmma_f32_16x16x32_f16(false, ax, false, bw, (short)0, ai,
                                                            false, false);
#pragma unroll
                for (int kb = 0; kb < 4; ++kb) {
                    v16h bh = load_b_sw(sWh, kb * 24 + ntg, lane);    // ntt=24
                    ahh = __builtin_amdgcn_wmma_f32_16x16x32_f16(false, ah[kb], false, bh,
                                                                 (short)0, ahh, false, false);
                }
                gi[g] = ai; gh[g] = ahh;
            }
            int rb = row0 + ((lane >> 4) << 3);
#pragma unroll
            for (int r = 0; r < 8; ++r) {
                float ir  = gi[0][r] + sbih[col];
                float hr  = gh[0][r] + sbhh[col];
                float iz  = gi[1][r] + sbih[HDIM + col];
                float hz  = gh[1][r] + sbhh[HDIM + col];
                float inn = gi[2][r] + sbih[2 * HDIM + col];
                float hn  = gh[2][r] + sbhh[2 * HDIM + col];
                float rg = sigmoid_f(ir + hr);
                float zg = sigmoid_f(iz + hz);
                float ng = tanh_f(inn + rg * hn);
                int row = rb + r;
                float hold = (float)hcur[row * HDIM + col];
                float hnew = (1.f - zg) * ng + zg * hold;
                hnxt[row * HDIM + col] = (_Float16)hnew;
            }
        }
        __syncthreads();

        // ctx[t] = h_new @ encW^T + enc_b  (256x128 @ 128x64), f16 to workspace
        _Float16* cd = ctx + (size_t)t * BATCH * CDIM;
        for (int task = wave; task < 64; task += NW) {
            int mt = task >> 2, ct = task & 3;
            int row0 = mt << 4, col0 = ct << 4;
            int col = col0 + (lane & 15);
            float bv = sencb[col];
            v8f acc;
#pragma unroll
            for (int r = 0; r < 8; ++r) acc[r] = bv;
#pragma unroll
            for (int kb = 0; kb < 4; ++kb) {
                v16h a = load_a_frag(hnxt, HDIM, row0, kb * 32, lane);
                v16h b = load_b_sw(seT, kb * 4 + ct, lane);           // ntt=4
                acc = __builtin_amdgcn_wmma_f32_16x16x32_f16(false, a, false, b, (short)0, acc,
                                                             false, false);
            }
            int rb = row0 + ((lane >> 4) << 3);
#pragma unroll
            for (int r = 0; r < 8; ++r)
                cd[(rb + r) * CDIM + col] = (_Float16)acc[r];
        }
        __syncthreads();
    }
}

// ----------------------------- Phase 2: SDE --------------------------------
extern "C" __global__ __launch_bounds__(512)
void sde_kernel(const float* __restrict__ xs, const float* __restrict__ ts,
                const float* __restrict__ noise,
                const float* __restrict__ fb1, const float* __restrict__ fb2,
                const float* __restrict__ fb3,
                const float* __restrict__ hb1, const float* __restrict__ hb2,
                const float* __restrict__ hb3, const float* __restrict__ hb4,
                const float* __restrict__ gb2,
                const _Float16* __restrict__ wfW1, const _Float16* __restrict__ wfW2,
                const _Float16* __restrict__ wfW3,
                const _Float16* __restrict__ whW1, const _Float16* __restrict__ whW2,
                const _Float16* __restrict__ whW3, const _Float16* __restrict__ whW4,
                const _Float16* __restrict__ wgW1, const _Float16* __restrict__ wgb1,
                const _Float16* __restrict__ wgW2,
                const _Float16* __restrict__ ctx,
                float* __restrict__ out) {
    extern __shared__ __attribute__((aligned(32))) char smem[];
    _Float16* sfW1 = (_Float16*)smem;          // 96 x 128  (swizzled)
    _Float16* sfW2 = sfW1 + 96 * HDIM;         // 128 x 128 (swizzled)
    _Float16* sfW3 = sfW2 + HDIM * HDIM;       // 128 x 16  (swizzled)
    _Float16* shW1 = sfW3 + HDIM * DIN;        // 32 x 128  (swizzled)
    _Float16* shW2 = shW1 + 32 * HDIM;         // 128 x 128 (swizzled)
    _Float16* shW3 = shW2 + HDIM * HDIM;       // 128 x 128 (swizzled)
    _Float16* shW4 = shW3 + HDIM * HDIM;       // 128 x 16  (swizzled)
    _Float16* sgW1 = shW4 + HDIM * DIN;        // 16 x 128
    _Float16* sgb1 = sgW1 + DIN * HDIM;        // 16 x 128
    _Float16* sgW2 = sgb1 + DIN * HDIM;        // 16 x 128
    _Float16* bufA = sgW2 + DIN * HDIM;        // 256 x 128
    _Float16* bufB = bufA + BATCH * HDIM;      // 256 x 128
    _Float16* sfd  = bufB + BATCH * HDIM;      // 256 x 16
    _Float16* shd  = sfd + BATCH * DIN;        // 256 x 16
    float* sz   = (float*)(shd + BATCH * DIN); // 256 x 16
    float* slq  = sz + BATCH * DIN;            // 256
    float* sfb1 = slq + BATCH;                 // 128
    float* sfb2 = sfb1 + HDIM;                 // 128
    float* sfb3 = sfb2 + HDIM;                 // 16
    float* shb1 = sfb3 + DIN;                  // 128
    float* shb2 = shb1 + HDIM;                 // 128
    float* shb3 = shb2 + HDIM;                 // 128
    float* shb4 = shb3 + HDIM;                 // 16
    float* sgb2 = shb4 + DIN;                  // 16

    const int tid = threadIdx.x, lane = tid & 31, wave = tid >> 5;
    const int NW = 16;

    copy_h8(wfW1, sfW1, 96 * HDIM, tid, 512);
    copy_h8(wfW2, sfW2, HDIM * HDIM, tid, 512);
    copy_h8(wfW3, sfW3, HDIM * DIN, tid, 512);
    copy_h8(whW1, shW1, 32 * HDIM, tid, 512);
    copy_h8(whW2, shW2, HDIM * HDIM, tid, 512);
    copy_h8(whW3, shW3, HDIM * HDIM, tid, 512);
    copy_h8(whW4, shW4, HDIM * DIN, tid, 512);
    copy_h8(wgW1, sgW1, DIN * HDIM, tid, 512);
    copy_h8(wgb1, sgb1, DIN * HDIM, tid, 512);
    copy_h8(wgW2, sgW2, DIN * HDIM, tid, 512);
    for (int i = tid; i < HDIM; i += 512) {
        sfb1[i] = fb1[i]; sfb2[i] = fb2[i];
        shb1[i] = hb1[i]; shb2[i] = hb2[i]; shb3[i] = hb3[i];
    }
    for (int i = tid; i < DIN; i += 512) {
        sfb3[i] = fb3[i]; shb4[i] = hb4[i]; sgb2[i] = gb2[i];
    }
    for (int i = tid; i < BATCH * DIN; i += 512) sz[i] = xs[i];  // z0 = xs[0]
    for (int i = tid; i < BATCH; i += 512) slq[i] = 0.f;
    for (int i = tid; i < BATCH * OUTD; i += 512) {              // zs[0]
        int b = i >> 3, k = i & 7;
        out[2 + i] = xs[b * DIN + 8 + k];
    }
    const float dt   = ts[1] - ts[0];
    const float sqdt = sqrtf(dt);
    __syncthreads();

    for (int i = 0; i < T_STEPS - 1; ++i) {
        // stage [c | z | 0] (K = 96) into bufA; c copy vectorized 8 halves at a time
        const _Float16* cp = ctx + (size_t)(i + 1) * BATCH * CDIM;  // i+1 <= 1023
        {
            const uint4* s = (const uint4*)cp;
            for (int idx = tid; idx < BATCH * (CDIM / 8); idx += 512) {
                int b = idx >> 3, c8 = idx & 7;
                *(uint4*)(bufA + b * HDIM + c8 * 8) = s[idx];
            }
        }
        for (int idx = tid; idx < BATCH * DIN; idx += 512) {
            int b = idx >> 4, d = idx & 15;
            bufA[b * HDIM + 64 + d] = (_Float16)sz[idx];
            bufA[b * HDIM + 80 + d] = (_Float16)0.f;
        }
        if (i + 2 < T_STEPS && tid < 256) {
            const char* p = (const char*)(ctx + (size_t)(i + 2) * BATCH * CDIM);
            __builtin_prefetch(p + tid * 128, 0, 1);
        }
        if (i + 1 < T_STEPS - 1 && tid < 256) {
            const char* p = (const char*)(noise + (size_t)(i + 1) * BATCH * DIN);
            __builtin_prefetch(p + tid * 64, 0, 1);
        }
        __syncthreads();

        // f-net
        gemm_layer<3, 1>(bufA, HDIM, 0, sfW1, HDIM, sfb1, bufB, HDIM, lane, wave, NW);
        __syncthreads();
        gemm_layer<4, 1>(bufB, HDIM, 0, sfW2, HDIM, sfb2, bufA, HDIM, lane, wave, NW);
        __syncthreads();
        gemm_layer<4, 0>(bufA, HDIM, 0, sfW3, DIN, sfb3, sfd, DIN, lane, wave, NW);
        // stage z padded to 32 into bufB (dead after f-L2)
        for (int idx = tid; idx < BATCH * DIN; idx += 512) {
            int b = idx >> 4, d = idx & 15;
            bufB[b * HDIM + d]      = (_Float16)sz[idx];
            bufB[b * HDIM + 16 + d] = (_Float16)0.f;
        }
        __syncthreads();

        // h-net
        gemm_layer<1, 1>(bufB, HDIM, 0, shW1, HDIM, shb1, bufA, HDIM, lane, wave, NW);
        __syncthreads();
        gemm_layer<4, 1>(bufA, HDIM, 0, shW2, HDIM, shb2, bufB, HDIM, lane, wave, NW);
        __syncthreads();
        gemm_layer<4, 1>(bufB, HDIM, 0, shW3, HDIM, shb3, bufA, HDIM, lane, wave, NW);
        __syncthreads();
        gemm_layer<4, 0>(bufA, HDIM, 0, shW4, DIN, shb4, shd, DIN, lane, wave, NW);
        __syncthreads();

        // g-net + state update: thread owns (b, 8 consecutive d)
        {
            int b = tid >> 1;
            int dbase = (tid & 1) << 3;
            const float* wp = noise + (size_t)i * BATCH * DIN + b * DIN;
            float lqp = 0.f;
            for (int j = 0; j < 8; ++j) {
                int d = dbase + j;
                float y = sz[b * DIN + d];
                const _Float16* g1 = sgW1 + d * HDIM;
                const _Float16* gB = sgb1 + d * HDIM;
                const _Float16* g2 = sgW2 + d * HDIM;
                float acc = 0.f;
                for (int h = 0; h < HDIM; h += 8) {
                    v8h a1 = *(const v8h*)(g1 + h);
                    v8h ab = *(const v8h*)(gB + h);
                    v8h a2 = *(const v8h*)(g2 + h);
#pragma unroll
                    for (int q = 0; q < 8; ++q)
                        acc += softplus_f(fmaf(y, (float)a1[q], (float)ab[q])) * (float)a2[q];
                }
                float gd  = softplus_f(acc + sgb2[d]);
                float fdv = (float)sfd[b * DIN + d];
                float hdv = (float)shd[b * DIN + d];
                float u   = (fdv - hdv) / (gd + 1e-7f);
                lqp += 0.5f * u * u * dt;
                float zn = y + fdv * dt + gd * sqdt * wp[d];
                sz[b * DIN + d] = zn;
                if (d >= 8)
                    out[2 + (size_t)(i + 1) * BATCH * OUTD + b * OUTD + (d - 8)] = zn;
            }
            atomicAdd(&slq[b], lqp);
        }
        __syncthreads();
    }

    if (tid == 0) {
        float s = 0.f;
        for (int b = 0; b < BATCH; ++b) s += slq[b];
        out[1] = s * (1.f / BATCH);  // logqp = mean(lq)
    }
}

// ------------------------- Phase 3: log-prob sum ---------------------------
extern "C" __global__ __launch_bounds__(256)
void logprob_kernel(const float* __restrict__ xs, const float* __restrict__ zsrc,
                    float* __restrict__ out) {
    __shared__ float red[256];
    int idx = blockIdx.x * 256 + threadIdx.x;  // < T*B*8 exactly
    int t = idx / (BATCH * OUTD);
    int r = idx - t * (BATCH * OUTD);
    int b = r >> 3, k = r & 7;
    float x = xs[(size_t)t * BATCH * DIN + b * DIN + 8 + k];
    float z = zsrc[2 + idx];
    float u = (x - z) * 100.f;  // / scale(0.01)
    float lp = -0.5f * u * u + (4.6051702f - 0.91893853f);  // -log(s) - 0.5*log(2pi)
    red[threadIdx.x] = lp;
    __syncthreads();
    for (int s = 128; s > 0; s >>= 1) {
        if (threadIdx.x < s) red[threadIdx.x] += red[threadIdx.x + s];
        __syncthreads();
    }
    if (threadIdx.x == 0) atomicAdd(out, red[0] * (1.f / BATCH));  // mean over b
}

// ------------------------------- launcher ----------------------------------
extern "C" void kernel_launch(void* const* d_in, const int* in_sizes, int n_in,
                              void* d_out, int out_size, void* d_ws, size_t ws_size,
                              hipStream_t stream) {
    (void)in_sizes; (void)n_in; (void)out_size; (void)ws_size;
    const float* xs    = (const float*)d_in[0];
    const float* ts    = (const float*)d_in[1];
    const float* noise = (const float*)d_in[2];
    const float* Wih   = (const float*)d_in[3];
    const float* Whh   = (const float*)d_in[4];
    const float* bih   = (const float*)d_in[5];
    const float* bhh   = (const float*)d_in[6];
    const float* encW  = (const float*)d_in[7];
    const float* encb  = (const float*)d_in[8];
    const float* fW1   = (const float*)d_in[9];
    const float* fb1   = (const float*)d_in[10];
    const float* fW2   = (const float*)d_in[11];
    const float* fb2   = (const float*)d_in[12];
    const float* fW3   = (const float*)d_in[13];
    const float* fb3   = (const float*)d_in[14];
    const float* hW1   = (const float*)d_in[15];
    const float* hb1   = (const float*)d_in[16];
    const float* hW2   = (const float*)d_in[17];
    const float* hb2   = (const float*)d_in[18];
    const float* hW3   = (const float*)d_in[19];
    const float* hb3   = (const float*)d_in[20];
    const float* hW4   = (const float*)d_in[21];
    const float* hb4   = (const float*)d_in[22];
    const float* gW1   = (const float*)d_in[23];
    const float* gb1   = (const float*)d_in[24];
    const float* gW2   = (const float*)d_in[25];
    const float* gb2   = (const float*)d_in[26];
    float* out = (float*)d_out;

    _Float16* ws   = (_Float16*)d_ws;
    _Float16* wWiT = ws;                    // 32*384
    _Float16* wWhT = wWiT + 32 * G3H;       // 128*384
    _Float16* weT  = wWhT + HDIM * G3H;     // 128*64
    _Float16* wfW1 = weT + HDIM * CDIM;     // 96*128
    _Float16* wfW2 = wfW1 + 96 * HDIM;      // 128*128
    _Float16* wfW3 = wfW2 + HDIM * HDIM;    // 128*16
    _Float16* whW1 = wfW3 + HDIM * DIN;     // 32*128
    _Float16* whW2 = whW1 + 32 * HDIM;      // 128*128
    _Float16* whW3 = whW2 + HDIM * HDIM;    // 128*128
    _Float16* whW4 = whW3 + HDIM * HDIM;    // 128*16
    _Float16* wgW1 = whW4 + HDIM * DIN;     // 16*128
    _Float16* wgb1 = wgW1 + DIN * HDIM;     // 16*128
    _Float16* wgW2 = wgb1 + DIN * HDIM;     // 16*128
    _Float16* wctx = wgW2 + DIN * HDIM;     // T*B*C f16 (~34 MB)

    auto nb = [](int n) { return (n + 255) / 256; };
    pack_b_frag<<<nb(32 * G3H), 256, 0, stream>>>(Wih, wWiT, G3H, DIN, 32);
    pack_b_frag<<<nb(HDIM * G3H), 256, 0, stream>>>(Whh, wWhT, G3H, HDIM, HDIM);
    pack_b_frag<<<nb(HDIM * CDIM), 256, 0, stream>>>(encW, weT, CDIM, HDIM, HDIM);
    pack_fW1_frag<<<nb(96 * HDIM), 256, 0, stream>>>(fW1, wfW1);
    pack_b_frag<<<nb(HDIM * HDIM), 256, 0, stream>>>(fW2, wfW2, HDIM, HDIM, HDIM);
    pack_b_frag<<<nb(HDIM * DIN), 256, 0, stream>>>(fW3, wfW3, DIN, HDIM, HDIM);
    pack_b_frag<<<nb(32 * HDIM), 256, 0, stream>>>(hW1, whW1, HDIM, DIN, 32);
    pack_b_frag<<<nb(HDIM * HDIM), 256, 0, stream>>>(hW2, whW2, HDIM, HDIM, HDIM);
    pack_b_frag<<<nb(HDIM * HDIM), 256, 0, stream>>>(hW3, whW3, HDIM, HDIM, HDIM);
    pack_b_frag<<<nb(HDIM * DIN), 256, 0, stream>>>(hW4, whW4, DIN, HDIM, HDIM);
    conv_f16_kernel<<<nb(DIN * HDIM), 256, 0, stream>>>(gW1, wgW1, DIN * HDIM);
    conv_f16_kernel<<<nb(DIN * HDIM), 256, 0, stream>>>(gb1, wgb1, DIN * HDIM);
    conv_f16_kernel<<<nb(DIN * HDIM), 256, 0, stream>>>(gW2, wgW2, DIN * HDIM);

    // Phase 1: persistent GRU workgroup. LDS = 286720 + 3328 bytes.
    size_t sh1 = (size_t)(32 * G3H + HDIM * G3H + HDIM * CDIM + BATCH * 32 +
                          2 * BATCH * HDIM) * 2 + (size_t)(2 * G3H + CDIM) * 4;
    gru_ctx_kernel<<<1, 512, sh1, stream>>>(xs, bih, bhh, encb, wWiT, wWhT, weT, wctx);

    hipMemsetAsync(d_out, 0, sizeof(float), stream);  // zero log_pxs accumulator

    // Phase 2: persistent SDE workgroup. LDS = 299008 + 20160 bytes (< 320KB WGP).
    size_t sh2 = (size_t)(96 * HDIM + 3 * HDIM * HDIM + 2 * HDIM * DIN + 32 * HDIM +
                          3 * DIN * HDIM + 2 * BATCH * HDIM + 2 * BATCH * DIN) * 2 +
                 (size_t)(BATCH * DIN + BATCH + 5 * HDIM + 3 * DIN) * 4;
    sde_kernel<<<1, 512, sh2, stream>>>(xs, ts, noise, fb1, fb2, fb3, hb1, hb2, hb3, hb4, gb2,
                                        wfW1, wfW2, wfW3, whW1, whW2, whW3, whW4,
                                        wgW1, wgb1, wgW2, wctx, out);

    // Phase 3: parallel Gaussian log-prob reduction over T*B*OUT elements.
    logprob_kernel<<<(T_STEPS * BATCH * OUTD) / 256, 256, 0, stream>>>(xs, out, out);
}
